// shallowPLRNN_61125974556827
// MI455X (gfx1250) — matrix operand order repaired and checked by reading, not compiled, not run on hardware
//
#include <hip/hip_runtime.h>
#include <hip/hip_bf16.h>
#include <stddef.h>

// ---------------- types ----------------
typedef __attribute__((ext_vector_type(16))) __bf16          v16bf;
typedef __attribute__((ext_vector_type(16))) unsigned short  v16us;
typedef __attribute__((ext_vector_type(8)))  unsigned short  v8us;
typedef __attribute__((ext_vector_type(8)))  float           v8f;

#define B_   256
#define T_   1024
#define DX_  64
#define DZ_  256
#define DH_  1024
#define S_   16
#define DF_  64
#define ALPHA_ 0.2f

// LDS padded strides (elements)
#define ZT32_S  260   // 256 fp32 + pad (row rotates 4 banks)
#define ZTBF_S  264   // 256 bf16 + pad
#define HTBF_S  1032  // 1024 bf16 + pad

// workspace byte offsets
#define OFF_G      0u
#define OFF_GINV   16384u
#define OFF_P64T   32768u      // 64x64 bf16
#define OFF_OBSP   40960u      // obs B-fragments: 4*8*32*16 bf16
#define OFF_ORDER  73728u      // 256 int
#define OFF_OFFS   74752u      // 17 int
#define OFF_W2P    131072u     // 16 * 64*8*32*16 bf16 = 8 MB
#define OFF_W1P    (131072u + 8388608u)  // 8 MB

__device__ __forceinline__ unsigned short f2bf(float x) {
  unsigned u = __float_as_uint(x);
  u += 0x7fffu + ((u >> 16) & 1u);       // round-to-nearest-even
  return (unsigned short)(u >> 16);
}
__device__ __forceinline__ float bf2f(unsigned short h) {
  return __uint_as_float(((unsigned)h) << 16);
}
// LDS fragment: two 16B vector loads, concat (16B-aligned addresses)
__device__ __forceinline__ v16us lds_frag(const unsigned short* p) {
  const v8us* q = (const v8us*)p;
  v8us a = q[0], b = q[1];
  return __builtin_shufflevector(a, b, 0, 1, 2, 3, 4, 5, 6, 7,
                                 8, 9, 10, 11, 12, 13, 14, 15);
}
// global packed fragment: 32B aligned by construction -> widest load
__device__ __forceinline__ v16us g_frag(const unsigned short* p) {
  return *(const v16us*)p;
}
__device__ __forceinline__ v8f wmma_bf16(v16us a, v16us b, v8f c) {
  return __builtin_amdgcn_wmma_f32_16x16x32_bf16(
      false, __builtin_bit_cast(v16bf, a),
      false, __builtin_bit_cast(v16bf, b),
      (short)0, c, false, false);
}

// ---------------- phase 0: G = obs^T obs (64x64) ----------------
__global__ __launch_bounds__(256) void k_gram(const float* __restrict__ obs,
                                              float* __restrict__ G) {
  int tid = threadIdx.x;
#pragma unroll 1
  for (int p = 0; p < 16; ++p) {
    int idx = tid * 16 + p;
    int i = idx >> 6, j = idx & 63;
    float s = 0.f;
    for (int k = 0; k < DZ_; ++k) s += obs[k * DX_ + i] * obs[k * DX_ + j];
    G[idx] = s;
  }
}

// ---------------- phase 1: Gauss-Jordan inverse of SPD 64x64 ----------------
__global__ __launch_bounds__(256) void k_invert(const float* __restrict__ G,
                                                float* __restrict__ Ginv) {
  __shared__ float Aug[64][130];
  __shared__ float fcol[64];
  int tid = threadIdx.x;
  for (int e = tid; e < 64 * 64; e += 256) Aug[e >> 6][e & 63] = G[e];
  for (int e = tid; e < 64 * 64; e += 256)
    Aug[e >> 6][64 + (e & 63)] = ((e >> 6) == (e & 63)) ? 1.f : 0.f;
  __syncthreads();
  for (int k = 0; k < 64; ++k) {
    if (tid < 64) fcol[tid] = Aug[tid][k];
    __syncthreads();
    float pv = 1.0f / fcol[k];
    if (tid < 128) Aug[k][tid] = Aug[k][tid] * pv;
    __syncthreads();
    for (int e = tid; e < 64 * 128; e += 256) {
      int i = e >> 7, c = e & 127;
      if (i != k) Aug[i][c] -= fcol[i] * Aug[k][c];
    }
    __syncthreads();
  }
  for (int e = tid; e < 4096; e += 256) Ginv[e] = Aug[e >> 6][64 + (e & 63)];
}

// ---------------- phase 2: P64T[j][i] bf16 (pinv cols :64, transposed) -------
__global__ __launch_bounds__(256) void k_p64t(const float* __restrict__ Ginv,
                                              const float* __restrict__ obs,
                                              unsigned short* __restrict__ p64t) {
  int tid = threadIdx.x;
#pragma unroll 1
  for (int p = 0; p < 16; ++p) {
    int idx = tid * 16 + p;
    int j = idx >> 6, i = idx & 63;
    float s = 0.f;
    for (int k = 0; k < 64; ++k) s += Ginv[i * 64 + k] * obs[j * DX_ + k];
    p64t[j * 64 + i] = f2bf(s);   // B^T layout: row=N(z dim), col=K(dx)
  }
}

// ---------------- phase 3: pack obs (B-frag) + W1/W2 (A-frag) to bf16 --------
// A-frag K map: Koff = (lane<16)?0:16; K = k0*32 + Koff + e   (contiguous)
// B-frag K map: Koff = (lane<16)?0:16; K = k0*32 + Koff + e   (contiguous)
#define N_OBSP (4 * 8 * 32 * 16)
#define N_W2P  (S_ * 64 * 8 * 32 * 16)
#define N_W1P  (S_ * 16 * 32 * 32 * 16)
__global__ __launch_bounds__(256) void k_pack(const float* __restrict__ obs,
                                              const float* __restrict__ W1t,
                                              const float* __restrict__ W2t,
                                              unsigned short* __restrict__ obsp,
                                              unsigned short* __restrict__ w1p,
                                              unsigned short* __restrict__ w2p) {
  const int total = N_OBSP + N_W2P + N_W1P;
  for (int idx = blockIdx.x * blockDim.x + threadIdx.x; idx < total;
       idx += gridDim.x * blockDim.x) {
    if (idx < N_OBSP) {  // obs: B operand (K=dz, N=dx)
      int t = idx;
      int e = t & 15; int lane = (t >> 4) & 31; int k0 = (t >> 9) & 7; int nt = t >> 12;
      int N = nt * 16 + (lane & 15);
      int K = k0 * 32 + ((lane >> 4) << 4) + e;
      obsp[t] = f2bf(obs[K * DX_ + N]);
    } else if (idx < N_OBSP + N_W2P) {  // W2: A operand (M=dh, K=dz)
      int t = idx - N_OBSP;
      int e = t & 15; int lane = (t >> 4) & 31; int k0 = (t >> 9) & 7;
      int mt = (t >> 12) & 63; int s = t >> 18;
      int m = mt * 16 + (lane & 15);
      int Koff = (lane >> 4) << 4;
      int k = k0 * 32 + Koff + e;
      w2p[t] = f2bf(W2t[((size_t)s * DH_ + m) * DZ_ + k]);
    } else {            // W1: A operand (M=dz, K=dh)
      int t = idx - N_OBSP - N_W2P;
      int e = t & 15; int lane = (t >> 4) & 31; int k0 = (t >> 9) & 31;
      int mt = (t >> 14) & 15; int s = t >> 18;
      int m = mt * 16 + (lane & 15);
      int Koff = (lane >> 4) << 4;
      int k = k0 * 32 + Koff + e;
      w1p[t] = f2bf(W1t[((size_t)s * DZ_ + m) * DH_ + k]);
    }
  }
}

// ---------------- phase 4: stable counting sort by subject ----------------
__global__ void k_group(const int* __restrict__ subject,
                        int* __restrict__ order, int* __restrict__ offs) {
  if (threadIdx.x != 0) return;
  int cnt[S_];
  for (int s = 0; s < S_; ++s) cnt[s] = 0;
  for (int b = 0; b < B_; ++b) cnt[subject[b]]++;
  int acc = 0;
  for (int s = 0; s < S_; ++s) { offs[s] = acc; acc += cnt[s]; cnt[s] = offs[s]; }
  offs[S_] = acc;
  for (int b = 0; b < B_; ++b) { int s = subject[b]; order[cnt[s]++] = b; }
}

// ---------------- phase 5: persistent recurrence, one 16-sample tile / block ----
__global__ __launch_bounds__(256) void k_recur(
    const float* __restrict__ x_gt,
    const float* __restrict__ A_table, const float* __restrict__ h1_table,
    const float* __restrict__ h2_table,
    const unsigned short* __restrict__ W1pack, const unsigned short* __restrict__ W2pack,
    const unsigned short* __restrict__ obs_pack, const unsigned short* __restrict__ p64t,
    const int* __restrict__ order, const int* __restrict__ offs,
    float* __restrict__ out) {
  __shared__ __attribute__((aligned(32))) float          ZT32[16][ZT32_S];
  __shared__ __attribute__((aligned(32))) unsigned short ZTbf[16][ZTBF_S];
  __shared__ __attribute__((aligned(32))) unsigned short HTbf[16][HTBF_S];
  __shared__ float As[DZ_], h1s[DZ_], h2s[DH_];
  __shared__ int bidx[16];

  const int tid  = threadIdx.x;
  const int lane = tid & 31;
  const int wave = tid >> 5;
  const int l16  = lane & 15;
  const int hi   = lane >> 4;            // 0 or 1

  const int s    = blockIdx.x >> 2;      // subject
  const int tile = blockIdx.x & 3;       // up to 64 samples / subject
  const int beg  = offs[s], end = offs[s + 1];
  const int base = beg + tile * 16;
  int nv = end - base; if (nv > 16) nv = 16;
  if (nv <= 0) return;

  if (tid < 16) bidx[tid] = order[(tid < nv) ? (base + tid) : base];
  if (tid < DZ_) { As[tid] = A_table[s * DZ_ + tid]; h1s[tid] = h1_table[s * DZ_ + tid]; }
  for (int i = tid; i < DH_; i += 256) h2s[i] = h2_table[s * DH_ + i];
  for (int i = tid; i < 16 * ZT32_S; i += 256) (&ZT32[0][0])[i] = 0.f;
  for (int i = tid; i < 16 * ZTBF_S; i += 256) (&ZTbf[0][0])[i] = 0;
  __syncthreads();

  const unsigned short* W2p = W2pack + (size_t)s * (64 * 8 * 32 * 16) + (size_t)lane * 16;
  const unsigned short* W1p = W1pack + (size_t)s * (16 * 32 * 32 * 16) + (size_t)lane * 16;

  // f_t = x_t @ P64 via WMMA (waves 0-3), then z[:,:64] = aeff*f + (1-aeff)*z
  auto tf_step = [&](int t, float aeff) {
    if (wave < 4) {
      const int nt = wave;
      const int b = bidx[l16];
      const float* xr = x_gt + ((size_t)b * T_ + t) * DX_;
      v8f facc = {};
#pragma unroll
      for (int kk = 0; kk < 2; ++kk) {
        const int k0 = kk * 32;
        v16us ua;
#pragma unroll
        for (int e = 0; e < 8; ++e) ua[e] = f2bf(xr[k0 + hi * 16 + e]);
#pragma unroll
        for (int e = 0; e < 8; ++e) ua[8 + e] = f2bf(xr[k0 + hi * 16 + 8 + e]);
        const unsigned short* bp = p64t + (nt * 16 + l16) * 64 + k0 + hi * 16;
        facc = wmma_bf16(ua, lds_frag(bp), facc);  // p64t row is 128B -> 32B aligned
      }
      const int j = nt * 16 + l16;       // forcing dim (<64)
      const float om = 1.f - aeff;
#pragma unroll
      for (int v = 0; v < 8; ++v) {
        const int samp = hi * 8 + v;
        float nz = aeff * facc[v] + om * ZT32[samp][j];
        ZT32[samp][j] = nz;
        ZTbf[samp][j] = f2bf(nz);
      }
    }
  };

  tf_step(0, 1.0f);                      // z0[:, :64] = f0
  __syncthreads();

  for (int t = 0; t < T_; ++t) {
    tf_step(t, ALPHA_);
    __syncthreads();

    // ---- GEMM1: hidden(1024x16) = relu(W2 @ z + h2) ----
    // k0 outer: load each Z B-fragment once, feed 8 accumulators (8 M-tiles/wave)
    {
      v8f acc[8];
#pragma unroll
      for (int i = 0; i < 8; ++i) acc[i] = (v8f){};
#pragma unroll
      for (int k0 = 0; k0 < 8; ++k0) {
        v16us bf = lds_frag(&ZTbf[l16][k0 * 32 + hi * 16]);
#pragma unroll
        for (int i = 0; i < 8; ++i) {
          const int mt = wave + i * 8;
          acc[i] = wmma_bf16(g_frag(W2p + (size_t)(mt * 8 + k0) * 512), bf, acc[i]);
        }
      }
#pragma unroll
      for (int i = 0; i < 8; ++i) {
        const int mt = wave + i * 8;
        const int mb = mt * 16 + hi * 8;
        v8us hv;
#pragma unroll
        for (int v = 0; v < 8; ++v) {
          float x = acc[i][v] + h2s[mb + v];
          hv[v] = f2bf(x > 0.f ? x : 0.f);
        }
        *(v8us*)&HTbf[l16][mb] = hv;
      }
    }
    __syncthreads();

    // ---- GEMM2: z(256x16) = A*z + W1 @ hidden + h1 ----
    // k0 outer: load each hidden B-fragment once, 2 accumulators (2 M-tiles/wave)
    {
      v8f acc0 = {}, acc1 = {};
#pragma unroll
      for (int k0 = 0; k0 < 32; ++k0) {
        v16us bf = lds_frag(&HTbf[l16][k0 * 32 + hi * 16]);
        acc0 = wmma_bf16(g_frag(W1p + (size_t)(wave * 32 + k0) * 512), bf, acc0);
        acc1 = wmma_bf16(g_frag(W1p + (size_t)((wave + 8) * 32 + k0) * 512), bf, acc1);
      }
#pragma unroll
      for (int i = 0; i < 2; ++i) {
        const v8f& a = i ? acc1 : acc0;
        const int mt = wave + i * 8;
        const int mb = mt * 16 + hi * 8;
#pragma unroll
        for (int v = 0; v < 8; ++v) {
          const int m = mb + v;
          float nz = As[m] * ZT32[l16][m] + a[v] + h1s[m];
          ZT32[l16][m] = nz;
          ZTbf[l16][m] = f2bf(nz);
        }
      }
    }
    __syncthreads();

    // ---- decode: out_t(16x64) = z^T @ obs ; waves 0-3, one N-tile each ----
    if (wave < 4) {
      const int nt = wave;
      v8f acc = {};
#pragma unroll
      for (int k0 = 0; k0 < 8; ++k0) {
        // A-frag from ZTbf: two contiguous 8-short chunks, both 16B aligned
        const v8us* zr = (const v8us*)&ZTbf[l16][k0 * 32 + hi * 16];
        v16us ua = __builtin_shufflevector(zr[0], zr[1], 0, 1, 2, 3, 4, 5, 6, 7,
                                           8, 9, 10, 11, 12, 13, 14, 15);
        acc = wmma_bf16(ua, g_frag(obs_pack + ((size_t)(nt * 8 + k0) * 32 + lane) * 16),
                        acc);
      }
      const int nx = nt * 16 + l16;
#pragma unroll
      for (int v = 0; v < 8; ++v) {
        const int samp = hi * 8 + v;
        if (samp < nv) {
          const int b = bidx[samp];
          out[((size_t)b * T_ + t) * DX_ + nx] = acc[v];
        }
      }
    }
    __syncthreads();
  }
}

// ---------------- host-side launch ----------------
extern "C" void kernel_launch(void* const* d_in, const int* in_sizes, int n_in,
                              void* d_out, int out_size, void* d_ws, size_t ws_size,
                              hipStream_t stream) {
  const float* x_gt   = (const float*)d_in[0];
  const int*   subj   = (const int*)d_in[1];
  const float* obs    = (const float*)d_in[2];
  const float* A_t    = (const float*)d_in[3];
  const float* W1_t   = (const float*)d_in[4];
  const float* W2_t   = (const float*)d_in[5];
  const float* h1_t   = (const float*)d_in[6];
  const float* h2_t   = (const float*)d_in[7];
  float* out = (float*)d_out;
  char* ws = (char*)d_ws;
  (void)in_sizes; (void)n_in; (void)out_size; (void)ws_size;

  float*          G    = (float*)(ws + OFF_G);
  float*          Ginv = (float*)(ws + OFF_GINV);
  unsigned short* p64t = (unsigned short*)(ws + OFF_P64T);
  unsigned short* obsp = (unsigned short*)(ws + OFF_OBSP);
  int*            ordr = (int*)(ws + OFF_ORDER);
  int*            offs = (int*)(ws + OFF_OFFS);
  unsigned short* w2p  = (unsigned short*)(ws + OFF_W2P);
  unsigned short* w1p  = (unsigned short*)(ws + OFF_W1P);

  hipLaunchKernelGGL(k_gram,   dim3(1),    dim3(256), 0, stream, obs, G);
  hipLaunchKernelGGL(k_invert, dim3(1),    dim3(256), 0, stream, G, Ginv);
  hipLaunchKernelGGL(k_p64t,   dim3(1),    dim3(256), 0, stream, Ginv, obs, p64t);
  hipLaunchKernelGGL(k_pack,   dim3(4096), dim3(256), 0, stream,
                     obs, W1_t, W2_t, obsp, w1p, w2p);
  hipLaunchKernelGGL(k_group,  dim3(1),    dim3(64),  0, stream, subj, ordr, offs);
  hipLaunchKernelGGL(k_recur,  dim3(64),   dim3(256), 0, stream,
                     x_gt, A_t, h1_t, h2_t, w1p, w2p, obsp, p64t, ordr, offs, out);
}
